// MultiHeadAttention_44178033607346
// MI455X (gfx1250) — matrix operand order repaired and checked by reading, not compiled, not run on hardware
//
#include <hip/hip_runtime.h>
#include <hip/hip_bf16.h>

// ---------------------------------------------------------------------------
// MHA forward for MI455X (gfx1250, wave32, WMMA f16 16x16x32 w/ f32 accum).
// Pipeline: cast/transpose prep -> QKV GEMM (WMMA, async-to-LDS double
// buffered) -> flash attention (WMMA, S^T trick avoids LDS transpose)
// -> output GEMM (WMMA).
// ---------------------------------------------------------------------------

typedef __attribute__((ext_vector_type(16))) _Float16 v16h;
typedef __attribute__((ext_vector_type(8)))  float    v8f;

#define D_MODEL   1024
#define NUM_HEADS 16
#define HEAD_DIM  64
#define SEQ       2048
#define BATCH     2
#define MTOT      (BATCH * SEQ)     // 4096 rows
#define N_QKV     (3 * D_MODEL)     // 3072

// ---- WMMA fragment loaders ------------------------------------------------
// A-operand (16xK=32 f16) per-lane layout: lanes 0-15 hold k=0..15,
// lanes 16-31 hold k=16..31 (contiguous, matching B).
__device__ __forceinline__ v16h load_frag_a(const _Float16* rowptr, int k_off, int lh) {
    union { v16h h; uint4 u[2]; } f;
    f.u[0] = *(const uint4*)(rowptr + k_off + 16 * lh);
    f.u[1] = *(const uint4*)(rowptr + k_off + 16 * lh + 8);
    return f.h;
}
// B-operand (K=32 x 16 f16) per-lane layout: lanes 0-15 hold k=0..15,
// lanes 16-31 hold k=16..31 (contiguous, same order as A).
__device__ __forceinline__ v16h load_frag_b(const _Float16* rowptr, int k_off, int lh) {
    union { v16h h; uint4 u[2]; } f;
    f.u[0] = *(const uint4*)(rowptr + k_off + 16 * lh);
    f.u[1] = *(const uint4*)(rowptr + k_off + 16 * lh + 8);
    return f.h;
}
__device__ __forceinline__ v8f wmma_f16(v16h a, v16h b, v8f c) {
    return __builtin_amdgcn_wmma_f32_16x16x32_f16(false, a, false, b, (short)0, c,
                                                  false, false);
}
__device__ __forceinline__ v8f vzero8() {
    v8f z = {0.f, 0.f, 0.f, 0.f, 0.f, 0.f, 0.f, 0.f};
    return z;
}

// ---- CDNA5 async memory->LDS copy (ASYNCcnt-tracked, no VGPR round trip) --
// Copies 32 bytes (two b128 lanes-worth) from gsrc to LDS at ldst.
// INST_OFFSET is added to BOTH the LDS and the global address (ISA 10.x).
__device__ __forceinline__ void async_copy32(const _Float16* gsrc, _Float16* ldst) {
    unsigned l = (unsigned)(uintptr_t)ldst;  // low 32 bits = LDS offset
    asm volatile("global_load_async_to_lds_b128 %0, %1, off"
                 :: "v"(l), "v"(gsrc) : "memory");
    asm volatile("global_load_async_to_lds_b128 %0, %1, off offset:16"
                 :: "v"(l), "v"(gsrc) : "memory");
}
__device__ __forceinline__ void wait_async0() {
    asm volatile("s_wait_asynccnt 0" ::: "memory");
}

// ---- Prep kernels ---------------------------------------------------------
__global__ void cast_f32_to_f16(const float* __restrict__ src,
                                _Float16* __restrict__ dst, int n) {
    int i = blockIdx.x * blockDim.x + threadIdx.x;
    if (i < n) dst[i] = (_Float16)src[i];
}

// dst[n][k] = (f16) src[k][n]   (src is [K][N] row-major)
__global__ void transpose_cast_f16(const float* __restrict__ src,
                                   _Float16* __restrict__ dst, int K, int N) {
    int n = blockIdx.x * 16 + threadIdx.x;
    int k = blockIdx.y * 16 + threadIdx.y;
    if (k < K && n < N) dst[(size_t)n * K + k] = (_Float16)src[(size_t)k * N + n];
}

// ---- Tiled WMMA GEMM: C[M,N] = A[M,K](f16) * Bt[N,K]^T(f16) + bias --------
// WG tile 128x128, 8 waves, wave tile 32x64 (2x4 subtiles of 16x16).
// Double-buffered LDS staged via global_load_async_to_lds_b128: stage s+1 is
// in flight while WMMAs consume stage s (one s_wait_asynccnt + one barrier
// per stage).
// MODE 0: QKV epilogue (scatter Q scaled, K, V-transposed, all f16).
// MODE 1: out-projection epilogue (fp32 + bias -> d_out).
template <int MODE>
__global__ __launch_bounds__(256) void gemm_wmma(
    const _Float16* __restrict__ A, const _Float16* __restrict__ Bt,
    const float* __restrict__ bias, int K, int N,
    _Float16* __restrict__ Qb, _Float16* __restrict__ Kb,
    _Float16* __restrict__ Vt, float* __restrict__ OutF) {
    constexpr int LDK = 40;  // padded LDS row stride (halves) to spread banks
    __shared__ _Float16 As[2][128 * LDK];
    __shared__ _Float16 Bs[2][128 * LDK];

    const int tid = threadIdx.x;
    const int lane = tid & 31, lane15 = lane & 15, lh = lane >> 4;
    const int w = tid >> 5;
    const int wm = (w >> 1) * 32;   // wave M offset in WG tile
    const int wn = (w & 1) * 64;    // wave N offset in WG tile
    const int m_wg = blockIdx.y * 128;
    const int n_wg = blockIdx.x * 128;

    v8f acc[2][4];
#pragma unroll
    for (int i = 0; i < 2; i++)
#pragma unroll
        for (int j = 0; j < 4; j++) acc[i][j] = vzero8();

    // cooperative staging: each thread owns 16 halves (32B) of one row half
    const int lrow = tid >> 1;
    const int lcol = (tid & 1) * 16;
    const _Float16* Agp = A + (size_t)(m_wg + lrow) * K + lcol;
    const _Float16* Bgp = Bt + (size_t)(n_wg + lrow) * K + lcol;
    const int lds_off = lrow * LDK + lcol;

    const int nstages = K >> 5;  // K/32
    // prologue: kick off stage 0
    async_copy32(Agp, &As[0][lds_off]);
    async_copy32(Bgp, &Bs[0][lds_off]);

    for (int s = 0; s < nstages; ++s) {
        wait_async0();     // this wave's in-flight stage has landed in LDS
        __syncthreads();   // ...and so has everyone else's
        if (s + 1 < nstages) {
            const int nb = (s + 1) & 1;
            async_copy32(Agp + (s + 1) * 32, &As[nb][lds_off]);
            async_copy32(Bgp + (s + 1) * 32, &Bs[nb][lds_off]);
        }
        const _Float16* curA = As[s & 1];
        const _Float16* curB = Bs[s & 1];

        v16h af[2], bf[4];
#pragma unroll
        for (int i = 0; i < 2; i++)
            af[i] = load_frag_a(curA + (wm + i * 16 + lane15) * LDK, 0, lh);
#pragma unroll
        for (int j = 0; j < 4; j++)
            bf[j] = load_frag_b(curB + (wn + j * 16 + lane15) * LDK, 0, lh);
#pragma unroll
        for (int i = 0; i < 2; i++)
#pragma unroll
            for (int j = 0; j < 4; j++) acc[i][j] = wmma_f16(af[i], bf[j], acc[i][j]);
        // next iteration's barrier fences re-use of this buffer
    }

    // epilogue: C/D layout -> lane holds n = lane15, rows m = r + 8*lh
#pragma unroll
    for (int i = 0; i < 2; i++) {
        const int mrow0 = m_wg + wm + i * 16 + 8 * lh;
#pragma unroll
        for (int j = 0; j < 4; j++) {
            const int n = n_wg + wn + j * 16 + lane15;
            const float bv = bias[n];
#pragma unroll
            for (int r = 0; r < 8; r++) {
                const int row = mrow0 + r;
                const float v = acc[i][j][r] + bv;
                if (MODE == 0) {
                    const int b = row >> 11, sidx = row & (SEQ - 1);
                    const int h = n / (3 * HEAD_DIM), off = n % (3 * HEAD_DIM);
                    const int bh = b * NUM_HEADS + h;
                    if (off < HEAD_DIM) {
                        // fold 1/sqrt(hd) into Q
                        Qb[((size_t)bh * SEQ + sidx) * HEAD_DIM + off] =
                            (_Float16)(v * 0.125f);
                    } else if (off < 2 * HEAD_DIM) {
                        Kb[((size_t)bh * SEQ + sidx) * HEAD_DIM + (off - HEAD_DIM)] =
                            (_Float16)v;
                    } else {
                        // V stored transposed [bh][hd][S] so it feeds the
                        // A-operand of O^T = V^T * P^T directly
                        Vt[((size_t)bh * HEAD_DIM + (off - 2 * HEAD_DIM)) * SEQ + sidx] =
                            (_Float16)v;
                    }
                } else {
                    OutF[(size_t)row * N + n] = v;
                }
            }
        }
    }
}

// ---- Flash attention ------------------------------------------------------
// One wave owns 16 queries of one (b,h); streams keys in blocks of 32.
// Computes S^T = K * Q^T (per-query softmax reduce = one shfl_xor(16)),
// then O^T += V^T * P^T where P^T already matches the WMMA B-operand lane
// layout after a single cross-half shuffle — no LDS transpose needed.
__global__ __launch_bounds__(128) void attention_fa(
    const _Float16* __restrict__ Qb, const _Float16* __restrict__ Kb,
    const _Float16* __restrict__ Vt, _Float16* __restrict__ Ov) {
    const int lane = threadIdx.x & 31, lane15 = lane & 15, lh = lane >> 4;
    const int w = threadIdx.x >> 5;
    const int bh = blockIdx.y;                     // 0..31
    const int q0 = blockIdx.x * 64 + w * 16;       // query tile base

    const _Float16* Qrow = Qb + ((size_t)bh * SEQ + q0 + lane15) * HEAD_DIM;
    const _Float16* Kbase = Kb + (size_t)bh * SEQ * HEAD_DIM;
    const _Float16* Vrow[4];
#pragma unroll
    for (int t = 0; t < 4; t++)
        Vrow[t] = Vt + ((size_t)bh * HEAD_DIM + t * 16 + lane15) * SEQ;

    v16h qf[2];
    qf[0] = load_frag_b(Qrow, 0, lh);
    qf[1] = load_frag_b(Qrow, 32, lh);

    v8f accO[4];
#pragma unroll
    for (int t = 0; t < 4; t++) accO[t] = vzero8();
    float m_i = -__builtin_inff();
    float l_i = 0.f;
    const float LOG2E = 1.44269504088896f;

    for (int kb = 0; kb < SEQ; kb += 32) {
        // S^T tiles: sc0 = keys kb..kb+15, sc1 = keys kb+16..kb+31 (m=key, n=query)
        v8f sc0 = vzero8(), sc1 = vzero8();
#pragma unroll
        for (int s = 0; s < 2; s++) {
            v16h ka0 = load_frag_a(Kbase + (size_t)(kb + lane15) * HEAD_DIM, s * 32, lh);
            v16h ka1 = load_frag_a(Kbase + (size_t)(kb + 16 + lane15) * HEAD_DIM, s * 32, lh);
            sc0 = wmma_f16(ka0, qf[s], sc0);
            sc1 = wmma_f16(ka1, qf[s], sc1);
        }
        // online softmax: lane pair (L, L^16) together covers all 32 keys of query n
        float mx = m_i;
#pragma unroll
        for (int r = 0; r < 8; r++) {
            mx = fmaxf(mx, sc0[r]);
            mx = fmaxf(mx, sc1[r]);
        }
        mx = fmaxf(mx, __shfl_xor(mx, 16));
        const float corr = exp2f((m_i - mx) * LOG2E);
        m_i = mx;

        float p0[8], p1[8], rs = 0.f;
#pragma unroll
        for (int r = 0; r < 8; r++) {
            p0[r] = exp2f((sc0[r] - mx) * LOG2E);
            p1[r] = exp2f((sc1[r] - mx) * LOG2E);
            rs += p0[r] + p1[r];
        }
        rs += __shfl_xor(rs, 16);
        l_i = l_i * corr + rs;
#pragma unroll
        for (int t = 0; t < 4; t++)
#pragma unroll
            for (int r = 0; r < 8; r++) accO[t][r] *= corr;

        // pack P^T into B-operand layout. Convert to f16 first and exchange a
        // packed pair per shuffle (8 ds_bpermute instead of 16):
        // lane lh=0 needs keys 0..15 (own sc0 + partner sc0),
        // lane lh=1 needs keys 16..31 (partner sc1 + own sc1).
        union { v16h h; _Float16 e[16]; } pb;
#pragma unroll
        for (int r = 0; r < 8; r++) {
            union { unsigned u; _Float16 h[2]; } own, oth;
            own.h[0] = (_Float16)p0[r];
            own.h[1] = (_Float16)p1[r];
            oth.u = __shfl_xor(own.u, 16);
            pb.e[r]     = lh ? oth.h[1] : own.h[0];
            pb.e[r + 8] = lh ? own.h[1] : oth.h[0];
        }
        // O^T(64x16) += V^T(64x32) * P^T(32x16)
#pragma unroll
        for (int t = 0; t < 4; t++) {
            v16h va = load_frag_a(Vrow[t], kb, lh);
            accO[t] = wmma_f16(va, pb.h, accO[t]);
        }
    }

    const float inv_l = 1.f / l_i;
    const int b = bh >> 4, h = bh & (NUM_HEADS - 1);
    const size_t base =
        ((size_t)b * SEQ + q0 + lane15) * D_MODEL + h * HEAD_DIM;
#pragma unroll
    for (int t = 0; t < 4; t++)
#pragma unroll
        for (int r = 0; r < 8; r++)
            Ov[base + t * 16 + r + 8 * lh] = (_Float16)(accO[t][r] * inv_l);
}

// ---- Host launcher --------------------------------------------------------
extern "C" void kernel_launch(void* const* d_in, const int* in_sizes, int n_in,
                              void* d_out, int out_size, void* d_ws, size_t ws_size,
                              hipStream_t stream) {
    const float* x    = (const float*)d_in[0];
    const float* Wqkv = (const float*)d_in[1];
    const float* bqkv = (const float*)d_in[2];
    const float* Wout = (const float*)d_in[3];
    const float* bout = (const float*)d_in[4];
    float* out = (float*)d_out;

    char* ws = (char*)d_ws;
    _Float16* x16   = (_Float16*)ws; ws += (size_t)MTOT * D_MODEL * 2;
    _Float16* WqkvT = (_Float16*)ws; ws += (size_t)N_QKV * D_MODEL * 2;
    _Float16* WoutT = (_Float16*)ws; ws += (size_t)D_MODEL * D_MODEL * 2;
    _Float16* Qb    = (_Float16*)ws; ws += (size_t)BATCH * NUM_HEADS * SEQ * HEAD_DIM * 2;
    _Float16* Kb    = (_Float16*)ws; ws += (size_t)BATCH * NUM_HEADS * SEQ * HEAD_DIM * 2;
    _Float16* Vt    = (_Float16*)ws; ws += (size_t)BATCH * NUM_HEADS * HEAD_DIM * SEQ * 2;
    _Float16* Val16 = (_Float16*)ws; ws += (size_t)MTOT * D_MODEL * 2;

    // 1) prep: cast x, transpose+cast weights (tiny vs GEMMs; L2-resident)
    const int nX = MTOT * D_MODEL;
    cast_f32_to_f16<<<nX / 256, 256, 0, stream>>>(x, x16, nX);
    dim3 tb(16, 16);
    transpose_cast_f16<<<dim3(N_QKV / 16, D_MODEL / 16), tb, 0, stream>>>(
        Wqkv, WqkvT, D_MODEL, N_QKV);
    transpose_cast_f16<<<dim3(D_MODEL / 16, D_MODEL / 16), tb, 0, stream>>>(
        Wout, WoutT, D_MODEL, D_MODEL);

    // 2) QKV projection: [4096,1024] x [1024,3072] -> scatter Q/K/V^T (f16)
    gemm_wmma<0><<<dim3(N_QKV / 128, MTOT / 128), 256, 0, stream>>>(
        x16, WqkvT, bqkv, D_MODEL, N_QKV, Qb, Kb, Vt, nullptr);

    // 3) flash attention per (b,h), 16 queries per wave
    attention_fa<<<dim3(SEQ / 64, BATCH * NUM_HEADS), 128, 0, stream>>>(
        Qb, Kb, Vt, Val16);

    // 4) output projection: [4096,1024] x [1024,1024] + bias -> fp32 out
    gemm_wmma<1><<<dim3(D_MODEL / 128, MTOT / 128), 256, 0, stream>>>(
        Val16, WoutT, bout, D_MODEL, D_MODEL, nullptr, nullptr, nullptr, out);
}
